// GCNModelWPathways_89541478187017
// MI455X (gfx1250) — matrix operand order repaired and checked by reading, not compiled, not run on hardware
//
#include <hip/hip_runtime.h>
#include <hip/hip_bf16.h>

typedef __attribute__((ext_vector_type(16))) _Float16 v16h;
typedef __attribute__((ext_vector_type(8)))  _Float16 v8h;
typedef __attribute__((ext_vector_type(8)))  float    v8f;

// ---------------- constants ----------------
#define NN   15135
#define BSZ  8
#define HF   64
#define EE   300000
#define RR   40000
#define NCMT 400
#define HFC  200
#define NCLS 2
#define DIM  192            // H * L
#define ROWF 512            // BSZ * HF, floats per node
#define NROWS  (NN*BSZ)     // 121080 gemm rows (valid)
#define NTILES ((NROWS + 15) / 16)   // 7568
#define NROWSP (NTILES * 16)         // 121088 padded rows

// ---------------- utility kernels ----------------
__global__ void fill_f32(float* p, float v, long n) {
    long i = (long)blockIdx.x * blockDim.x + threadIdx.x;
    if (i < n) p[i] = v;
}

__global__ void f32_to_f16(const float* __restrict__ s, _Float16* __restrict__ d, int n) {
    int i = blockIdx.x * blockDim.x + threadIdx.x;
    if (i < n) d[i] = (_Float16)s[i];
}

// x[b][n][f] -> hh[(n*BSZ+b)*64 + f] as f16
__global__ void x_to_half(const float* __restrict__ x, _Float16* __restrict__ hh) {
    long i = (long)blockIdx.x * blockDim.x + threadIdx.x;
    if (i >= (long)NROWS * HF) return;
    long n   = i >> 9;          // /512
    int  rem = (int)(i & 511);
    int  b   = rem >> 6;
    int  f   = rem & 63;
    hh[i] = (_Float16)x[((long)b * NN + n) * HF + f];
}

// ---------------- degree / norm ----------------
__global__ void deg_acc(const int* __restrict__ dst, float* deg, int e) {
    int i = blockIdx.x * blockDim.x + threadIdx.x;
    if (i < e) atomicAdd(&deg[dst[i]], 1.0f);
}
__global__ void rsqrt_k(float* d, int n) {
    int i = blockIdx.x * blockDim.x + threadIdx.x;
    if (i < n) d[i] = rsqrtf(d[i]);   // deg >= 1 (self loop) so no zero check
}

// ---------------- WMMA GEMM: out[row][h] = sum_k A[row][k] * W[h][k] ----------------
// One wave -> 16 rows x 64 cols.  blockDim = 256 (8 waves).
// A/out are padded to NROWSP rows, grid covers tiles exactly -> completely branch-free,
// EXEC all-1s at every WMMA. Pad rows produce garbage that is never read downstream.
__global__ void gemm64_wmma(const _Float16* __restrict__ A, const _Float16* __restrict__ W,
                            float* __restrict__ out) {
    const int lane = threadIdx.x & 31;
    const int wave = threadIdx.x >> 5;
    const int tile = blockIdx.x * 8 + wave;
    const int rowbase = tile * 16;
    const int half = lane >> 4;              // 0: lanes 0-15, 1: lanes 16-31
    const int l16  = lane & 15;

    // B: lane holds column c*16+l16 of output = row of W (contiguous f16).
    // lanes 0-15 -> K = ks*32 + 0..15 ; lanes 16-31 -> K = ks*32 + 16..31
    v16h Bv[4][2];
#pragma unroll
    for (int c = 0; c < 4; ++c) {
        const int coln = c * 16 + l16;
#pragma unroll
        for (int ks = 0; ks < 2; ++ks)
            Bv[c][ks] = *(const v16h*)(W + coln * HF + ks * 32 + half * 16);
    }

    // A: lane<16 row=rowbase+l16 holds K {kb..kb+15}; lane>=16 holds {kb+16..kb+31}
    const _Float16* ap = A + (long)(rowbase + l16) * HF;
    v16h Av[2];
#pragma unroll
    for (int ks = 0; ks < 2; ++ks) {
        const int k0 = ks * 32 + half * 16;
        v8h lo = *(const v8h*)(ap + k0);
        v8h hi = *(const v8h*)(ap + k0 + 8);
        v16h a;
#pragma unroll
        for (int i = 0; i < 8; ++i) { a[i] = lo[i]; a[i + 8] = hi[i]; }
        Av[ks] = a;
    }

    v8f acc[4] = {};
#pragma unroll
    for (int ks = 0; ks < 2; ++ks)
#pragma unroll
        for (int c = 0; c < 4; ++c)
            acc[c] = __builtin_amdgcn_wmma_f32_16x16x32_f16(
                false, Av[ks], false, Bv[c][ks], (short)0, acc[c], false, false);

    // C layout: lane -> col = l16, VGPR v -> row = rowbase + v + half*8 (unguarded, padded)
#pragma unroll
    for (int c = 0; c < 4; ++c) {
#pragma unroll
        for (int v = 0; v < 8; ++v) {
            const long r = rowbase + v + half * 8;
            out[r * HF + c * 16 + l16] = acc[c][v];
        }
    }
}

// ---------------- self-loop init: h[n] = m[n] * dinv[n]^2 (plain stores) ----------------
__global__ void init_self(const float* __restrict__ m, const float* __restrict__ dinv,
                          float* __restrict__ out) {
    long i = (long)blockIdx.x * blockDim.x + threadIdx.x;
    if (i >= (long)NN * ROWF) return;
    const int n = (int)(i >> 9);
    const float w = dinv[n];
    out[i] = m[i] * w * w;
}

// ---------------- edge scatter: h[dst] += m[src] * dinv[src]*dinv[dst] ----------------
// 128 threads per edge, 4 floats per thread (float4 load + 4 f32 atomics)
__global__ void scatter_edges(const int* __restrict__ ei, const float* __restrict__ m,
                              const float* __restrict__ dinv, float* __restrict__ out) {
    long t = (long)blockIdx.x * blockDim.x + threadIdx.x;
    const int e  = (int)(t >> 7);
    if (e >= EE) return;
    const int j4 = (int)(t & 127) * 4;
    const int s = ei[e];
    const int d = ei[EE + e];
    const float w = dinv[s] * dinv[d];
    const float4 mv = *(const float4*)(m + (long)s * ROWF + j4);
    float* o = out + (long)d * ROWF + j4;
    atomicAdd(o + 0, mv.x * w);
    atomicAdd(o + 1, mv.y * w);
    atomicAdd(o + 2, mv.z * w);
    atomicAdd(o + 3, mv.w * w);
}

// ---------------- bias + relu (in place) + f16 copy for next layer ----------------
__global__ void bias_relu_half(float* __restrict__ h, _Float16* __restrict__ hh,
                               const float* __restrict__ b) {
    long i = (long)blockIdx.x * blockDim.x + threadIdx.x;
    if (i >= (long)NN * ROWF) return;
    const int f = (int)(i & 63);
    float v = h[i] + b[f];
    v = fmaxf(v, 0.0f);
    h[i]  = v;
    hh[i] = (_Float16)v;
}

// ---------------- pathway readout: sums[col] += hc[row] ----------------
// hc feature d = feat*3 + layer; 1536 = BSZ*DIM threads per index
__global__ void readout_scatter(const int* __restrict__ row, const int* __restrict__ col,
                                const float* __restrict__ h1, const float* __restrict__ h2,
                                const float* __restrict__ h3, float* __restrict__ sums) {
    long t = (long)blockIdx.x * blockDim.x + threadIdx.x;
    const int i = (int)(t / (BSZ * DIM));
    if (i >= RR) return;
    const int d = (int)(t % (BSZ * DIM));
    const int r = row[i];
    const int c = col[i];
    const int b   = d / DIM;
    const int rem = d % DIM;
    const int f   = rem / 3;
    const int l   = rem % 3;
    const float* h = (l == 0) ? h1 : (l == 1) ? h2 : h3;
    const float v = h[((long)r * BSZ + b) * HF + f];
    atomicAdd(&sums[(long)c * (BSZ * DIM) + d], v);
}

__global__ void cnt_acc(const int* __restrict__ col, float* cnt) {
    int i = blockIdx.x * blockDim.x + threadIdx.x;
    if (i < RR) atomicAdd(&cnt[col[i]], 1.0f);
}

// ---------------- head: mean -> fc -> lin1(relu) -> lin2 -> log_softmax ----------------
__global__ void head(const float* __restrict__ sums, const float* __restrict__ cnt,
                     const float* __restrict__ fcw, const float* __restrict__ fcb,
                     const float* __restrict__ l1w, const float* __restrict__ l1b,
                     const float* __restrict__ l2w, const float* __restrict__ l2b,
                     float* __restrict__ out) {
    __shared__ float s[BSZ * NCMT];    // 3200
    __shared__ float z1[BSZ * HFC];    // 1600
    const int tid = threadIdx.x;

    for (int idx = tid; idx < BSZ * NCMT; idx += 256) {
        const int b = idx / NCMT, c = idx % NCMT;
        const float* p = sums + (long)c * (BSZ * DIM) + b * DIM;
        float acc = 0.0f;
        for (int d = 0; d < DIM; ++d) acc += p[d] * fcw[d];
        const float ct = fmaxf(cnt[c], 1.0f);
        s[b * NCMT + c] = acc / ct + fcb[0];
    }
    __syncthreads();

    for (int idx = tid; idx < BSZ * HFC; idx += 256) {
        const int b = idx / HFC, o = idx % HFC;
        float acc = l1b[o];
        const float* wrow = l1w + o * NCMT;
        const float* srow = s + b * NCMT;
        for (int c = 0; c < NCMT; ++c) acc += srow[c] * wrow[c];
        z1[b * HFC + o] = fmaxf(acc, 0.0f);
    }
    __syncthreads();

    if (tid < BSZ) {
        const int b = tid;
        float z[NCLS];
        for (int o = 0; o < NCLS; ++o) {
            float acc = l2b[o];
            const float* wrow = l2w + o * HFC;
            const float* zrow = z1 + b * HFC;
            for (int c = 0; c < HFC; ++c) acc += zrow[c] * wrow[c];
            z[o] = acc;
        }
        const float mx  = fmaxf(z[0], z[1]);
        const float lse = mx + logf(expf(z[0] - mx) + expf(z[1] - mx));
        out[b * NCLS + 0] = z[0] - lse;
        out[b * NCLS + 1] = z[1] - lse;
    }
}

// ---------------- launcher ----------------
extern "C" void kernel_launch(void* const* d_in, const int* in_sizes, int n_in,
                              void* d_out, int out_size, void* d_ws, size_t ws_size,
                              hipStream_t stream) {
    const float* x    = (const float*)d_in[0];
    const int*   edge = (const int*)d_in[2];     // [2, E]
    const int*   row  = (const int*)d_in[3];
    const int*   col  = (const int*)d_in[4];
    const float* W[3]  = { (const float*)d_in[5], (const float*)d_in[7], (const float*)d_in[9] };
    const float* bv[3] = { (const float*)d_in[6], (const float*)d_in[8], (const float*)d_in[10] };
    const float* fcw = (const float*)d_in[11];
    const float* fcb = (const float*)d_in[12];
    const float* l1w = (const float*)d_in[13];
    const float* l1b = (const float*)d_in[14];
    const float* l2w = (const float*)d_in[15];
    const float* l2b = (const float*)d_in[16];
    float* out = (float*)d_out;

    const long NEL  = (long)NROWS  * HF;         // 7,749,120 valid elements
    const long NELP = (long)NROWSP * HF;         // padded to tile boundary

    // carve workspace (256B-aligned)
    char* p = (char*)d_ws;
    auto carve = [&](size_t bytes) -> char* {
        char* q = p; p += (bytes + 255) & ~(size_t)255; return q;
    };
    _Float16* hh   = (_Float16*)carve(NELP * 2);   // padded
    _Float16* wh   = (_Float16*)carve(HF * HF * 2);
    float*    m    = (float*)carve(NELP * 4);      // padded
    float*    h1   = (float*)carve(NEL * 4);
    float*    h2   = (float*)carve(NEL * 4);
    float*    h3   = (float*)carve(NEL * 4);
    float*    dinv = (float*)carve(NN * 4);
    float*    sums = (float*)carve((size_t)NCMT * BSZ * DIM * 4);
    float*    cnt  = (float*)carve(NCMT * 4);
    float*    Hs[3] = { h1, h2, h3 };

    const int TB = 256;
    const int gNEL  = (int)((NEL + TB - 1) / TB);                  // elementwise over N*BS*64
    const int gN    = (NN + TB - 1) / TB;
    const int gE    = (EE + TB - 1) / TB;
    const int gR    = (RR + TB - 1) / TB;
    const int gGemm = NTILES / 8;                                  // 7568/8 = 946, exact
    const int gScat = (int)(((long)EE * 128 + TB - 1) / TB);       // 128 thr/edge
    const int gRead = (int)(((long)RR * (BSZ * DIM) + TB - 1) / TB);

    // degree / normalization (dinv doubles as deg, init 1.0 for self loop)
    fill_f32<<<gN, TB, 0, stream>>>(dinv, 1.0f, NN);
    deg_acc<<<gE, TB, 0, stream>>>(edge + EE, dinv, EE);
    rsqrt_k<<<gN, TB, 0, stream>>>(dinv, NN);

    // h0 (f16, [N][BS][64])
    x_to_half<<<gNEL, TB, 0, stream>>>(x, hh);

    for (int l = 0; l < 3; ++l) {
        f32_to_f16<<<(HF * HF + TB - 1) / TB, TB, 0, stream>>>(W[l], wh, HF * HF);
        gemm64_wmma<<<gGemm, TB, 0, stream>>>(hh, wh, m);
        init_self<<<gNEL, TB, 0, stream>>>(m, dinv, Hs[l]);
        scatter_edges<<<gScat, TB, 0, stream>>>(edge, m, dinv, Hs[l]);
        bias_relu_half<<<gNEL, TB, 0, stream>>>(Hs[l], hh, bv[l]);
    }

    // readout
    fill_f32<<<(int)(((long)NCMT * BSZ * DIM + TB - 1) / TB), TB, 0, stream>>>(
        sums, 0.0f, (long)NCMT * BSZ * DIM);
    fill_f32<<<(NCMT + TB - 1) / TB, TB, 0, stream>>>(cnt, 0.0f, NCMT);
    cnt_acc<<<gR, TB, 0, stream>>>(col, cnt);
    readout_scatter<<<gRead, TB, 0, stream>>>(row, col, h1, h2, h3, sums);

    head<<<1, TB, 0, stream>>>(sums, cnt, fcw, fcb, l1w, l1b, l2w, l2b, out);
}
